// MultiHeadAttention_61426622267521
// MI455X (gfx1250) — matrix operand order repaired and checked by reading, not compiled, not run on hardware
//
#include <hip/hip_runtime.h>
#include <hip/hip_bf16.h>

typedef __attribute__((ext_vector_type(16))) __bf16 v16bf;
typedef __attribute__((ext_vector_type(8)))  __bf16 v8bf;
typedef __attribute__((ext_vector_type(8)))  float  v8f;

constexpr int kT  = 512;                 // sequence length
constexpr int kD  = 256;                 // hidden per head
constexpr int kH  = 8;                   // heads
constexpr int cB  = 32;                  // batch
constexpr int kF  = kH * kD;             // 2048 concat features
constexpr int BM  = 32;                  // query rows per workgroup
constexpr int PROJ_ELEMS = cB * kT * kD; // 4,194,304
constexpr int ASUM_ELEMS = cB * kT * kT; // 8,388,608

static __device__ inline v8f vzero8() {
  v8f z;
#pragma unroll
  for (int i = 0; i < 8; ++i) z[i] = 0.0f;
  return z;
}

static __device__ inline v16bf cat16(v8bf a, v8bf b) {
  return __builtin_shufflevector(a, b, 0, 1, 2, 3, 4, 5, 6, 7,
                                       8, 9, 10, 11, 12, 13, 14, 15);
}

static __device__ inline __bf16 f2bf(float x) { return (__bf16)x; }

// ---------------------------------------------------------------------------
// Kernel 0: zero the A_sum accumulation region (harness poisons d_out).
// ---------------------------------------------------------------------------
__global__ __launch_bounds__(256) void zero_f4(float4* __restrict__ p, int n4) {
  int i = blockIdx.x * 256 + threadIdx.x;
  if (i < n4) p[i] = make_float4(0.f, 0.f, 0.f, 0.f);
}

// ---------------------------------------------------------------------------
// Kernel 1: attention for one (head*batch) b and one 32-row query tile.
//   S = Q K^T (bf16 WMMA, f32 acc, held in registers: 8 x v8f per wave)
//   masked softmax (shuffle + LDS reduction), A_sum atomics,
//   O = P V (bf16 WMMA), O written as bf16 into X[batch][t][h*256+d].
// ---------------------------------------------------------------------------
__global__ __launch_bounds__(256) void attn_kernel(
    const float* __restrict__ q, const float* __restrict__ kmat,
    const float* __restrict__ vmat, const unsigned char* __restrict__ mask,
    __bf16* __restrict__ X, float* __restrict__ asum) {
  __shared__ __align__(128) unsigned char smem[65536];
  __bf16* sQ  = (__bf16*)smem;             // [32][256] bf16 (phase 1)
  __bf16* sK  = (__bf16*)(smem + 16384);   // [64][256] bf16 (phase 1)
  __bf16* sP  = (__bf16*)smem;             // [32][512] bf16 (phase 2->3)
  __bf16* sVT = (__bf16*)(smem + 32768);   // [256][64] bf16 (phase 3)
  float* sRedM = (float*)(smem + 49152);           // [32][4]
  float* sRowM = (float*)(smem + 49152 + 512);     // [32]
  float* sRedL = (float*)(smem + 49152 + 1024);    // [32][4]
  float* sRowL = (float*)(smem + 49152 + 1536);    // [32]

  const int qt = blockIdx.x;        // 0..15
  const int b  = blockIdx.y;        // 0..255 (= head*32 + batch)
  const int t0 = qt * BM;
  const int head = b >> 5, batch = b & 31;
  const int tid  = threadIdx.x;
  const int wave = tid >> 5, lane = tid & 31;
  const int ln = lane & 15, hf = lane >> 4;   // C-layout: lane half
  const int mi = wave >> 2, ni = wave & 3;    // S-phase fragment (m,n) tile

  // ---- Phase 0: stage Q tile (f32 -> bf16) -------------------------------
  {
    const int m = tid >> 3, d0 = (tid & 7) * 32;
    const float4* src = (const float4*)(q + ((size_t)b * kT + t0 + m) * kD + d0);
#pragma unroll
    for (int j = 0; j < 8; ++j) {
      float4 f = src[j];
      int o = m * 256 + d0 + j * 4;
      sQ[o + 0] = f2bf(f.x); sQ[o + 1] = f2bf(f.y);
      sQ[o + 2] = f2bf(f.z); sQ[o + 3] = f2bf(f.w);
    }
  }

  // ---- Phase 1: S = Q K^T, accumulators in registers ---------------------
  v8f accS[8];
#pragma unroll
  for (int sc = 0; sc < 8; ++sc) accS[sc] = vzero8();

#pragma unroll
  for (int sc = 0; sc < 8; ++sc) {
    const int s0 = sc * 64;
    {  // stage 64 keys x 256 dims (f32 -> bf16)
      const int i = tid >> 2, d0k = (tid & 3) * 64;
      const float4* src =
          (const float4*)(kmat + ((size_t)b * kT + s0 + i) * kD + d0k);
#pragma unroll
      for (int j = 0; j < 16; ++j) {
        float4 f = src[j];
        int o = i * 256 + d0k + j * 4;
        sK[o + 0] = f2bf(f.x); sK[o + 1] = f2bf(f.y);
        sK[o + 2] = f2bf(f.z); sK[o + 3] = f2bf(f.w);
      }
    }
    __syncthreads();
    const int mA = mi * 16 + ln;   // A fragment row
    const int sB = ni * 16 + ln;   // B fragment column (key index)
#pragma unroll
    for (int kd = 0; kd < 8; ++kd) {
      const int kA = kd * 32 + hf * 8;
      v8bf a0 = *(const v8bf*)(sQ + mA * 256 + kA);
      v8bf a1 = *(const v8bf*)(sQ + mA * 256 + kA + 16);
      const int kB2 = kd * 32 + hf * 8;
      v8bf b0 = *(const v8bf*)(sK + sB * 256 + kB2);
      v8bf b1 = *(const v8bf*)(sK + sB * 256 + kB2 + 16);
      accS[sc] = __builtin_amdgcn_wmma_f32_16x16x32_bf16(
          false, cat16(a0, a1), false, cat16(b0, b1), (short)0, accS[sc],
          false, false);
    }
    __syncthreads();
  }

  // ---- Phase 2: mask + softmax over full rows (512 cols in regs) ---------
  const int rowBase = t0 + mi * 16 + hf * 8;
  float mx[8];
#pragma unroll
  for (int r = 0; r < 8; ++r) mx[r] = -__builtin_inff();
#pragma unroll
  for (int sc = 0; sc < 8; ++sc) {
#pragma unroll
    for (int r = 0; r < 8; ++r) {
      const int row = rowBase + r;
      const int col = sc * 64 + ni * 16 + ln;
      float s = accS[sc][r];
      if (mask[((size_t)b * kT + row) * kT + col]) s = -__builtin_inff();
      accS[sc][r] = s;
      mx[r] = fmaxf(mx[r], s);
    }
  }
#pragma unroll
  for (int r = 0; r < 8; ++r) {
    mx[r] = fmaxf(mx[r], __shfl_xor(mx[r], 8, 16));
    mx[r] = fmaxf(mx[r], __shfl_xor(mx[r], 4, 16));
    mx[r] = fmaxf(mx[r], __shfl_xor(mx[r], 2, 16));
    mx[r] = fmaxf(mx[r], __shfl_xor(mx[r], 1, 16));
  }
  if (ln == 0) {
#pragma unroll
    for (int r = 0; r < 8; ++r)
      sRedM[(mi * 16 + hf * 8 + r) * 4 + ni] = mx[r];
  }
  __syncthreads();
  if (tid < 32) {
    float a = fmaxf(fmaxf(sRedM[tid * 4 + 0], sRedM[tid * 4 + 1]),
                    fmaxf(sRedM[tid * 4 + 2], sRedM[tid * 4 + 3]));
    sRowM[tid] = a;
  }
  __syncthreads();
  float Mr[8], sm[8];
#pragma unroll
  for (int r = 0; r < 8; ++r) {
    Mr[r] = sRowM[mi * 16 + hf * 8 + r];
    sm[r] = 0.f;
  }
#pragma unroll
  for (int sc = 0; sc < 8; ++sc) {
#pragma unroll
    for (int r = 0; r < 8; ++r) {
      float e = (Mr[r] == -__builtin_inff()) ? 0.f
                                             : __expf(accS[sc][r] - Mr[r]);
      accS[sc][r] = e;
      sm[r] += e;
    }
  }
#pragma unroll
  for (int r = 0; r < 8; ++r) {
    sm[r] += __shfl_xor(sm[r], 8, 16);
    sm[r] += __shfl_xor(sm[r], 4, 16);
    sm[r] += __shfl_xor(sm[r], 2, 16);
    sm[r] += __shfl_xor(sm[r], 1, 16);
  }
  if (ln == 0) {
#pragma unroll
    for (int r = 0; r < 8; ++r)
      sRedL[(mi * 16 + hf * 8 + r) * 4 + ni] = sm[r];
  }
  __syncthreads();
  if (tid < 32) {
    sRowL[tid] = sRedL[tid * 4 + 0] + sRedL[tid * 4 + 1] +
                 sRedL[tid * 4 + 2] + sRedL[tid * 4 + 3];
  }
  __syncthreads();
  float rinv[8];
#pragma unroll
  for (int r = 0; r < 8; ++r) {
    float l = sRowL[mi * 16 + hf * 8 + r];
    rinv[r] = (l > 0.f) ? 1.f / l : 0.f;
  }
  // normalized probabilities: A_sum atomics + bf16 P into LDS
#pragma unroll
  for (int sc = 0; sc < 8; ++sc) {
#pragma unroll
    for (int r = 0; r < 8; ++r) {
      const int row = rowBase + r;
      const int col = sc * 64 + ni * 16 + ln;
      float p = accS[sc][r] * rinv[r];
      atomicAdd(asum + ((size_t)batch * kT + row) * kT + col, p);
      sP[(mi * 16 + hf * 8 + r) * 512 + col] = f2bf(p);
    }
  }
  __syncthreads();

  // ---- Phase 3: O = P V ---------------------------------------------------
  v8f accO[4];
#pragma unroll
  for (int jj = 0; jj < 4; ++jj) accO[jj] = vzero8();

  for (int vc = 0; vc < 8; ++vc) {
    const int s0 = vc * 64;
    {  // stage V chunk transposed: sVT[d][s_local] (f32 -> bf16)
      const int i = tid & 63, d0v = (tid >> 6) * 64;
      const float4* src =
          (const float4*)(vmat + ((size_t)b * kT + s0 + i) * kD + d0v);
#pragma unroll
      for (int j = 0; j < 16; ++j) {
        float4 f = src[j];
        int dd = d0v + j * 4;
        sVT[(dd + 0) * 64 + i] = f2bf(f.x);
        sVT[(dd + 1) * 64 + i] = f2bf(f.y);
        sVT[(dd + 2) * 64 + i] = f2bf(f.z);
        sVT[(dd + 3) * 64 + i] = f2bf(f.w);
      }
    }
    __syncthreads();
    const int mA = mi * 16 + ln;
#pragma unroll
    for (int kkstep = 0; kkstep < 2; ++kkstep) {
      const int kg = s0 + kkstep * 32;  // global s (columns of P)
      const int kl = kkstep * 32;       // local s within sVT
      v8bf a0 = *(const v8bf*)(sP + mA * 512 + kg + hf * 8);
      v8bf a1 = *(const v8bf*)(sP + mA * 512 + kg + hf * 8 + 16);
      v16bf Af = cat16(a0, a1);
#pragma unroll
      for (int jj = 0; jj < 4; ++jj) {
        const int nj = (wave & 3) * 4 + jj;
        const int dcol = nj * 16 + ln;
        v8bf b0 = *(const v8bf*)(sVT + dcol * 64 + kl + hf * 8);
        v8bf b1 = *(const v8bf*)(sVT + dcol * 64 + kl + hf * 8 + 16);
        accO[jj] = __builtin_amdgcn_wmma_f32_16x16x32_bf16(
            false, Af, false, cat16(b0, b1), (short)0, accO[jj], false, false);
      }
    }
    __syncthreads();
  }

  // write O as bf16 into X[batch][t][head*256 + d]
#pragma unroll
  for (int jj = 0; jj < 4; ++jj) {
    const int nj = (wave & 3) * 4 + jj;
#pragma unroll
    for (int r = 0; r < 8; ++r) {
      const int trow = t0 + mi * 16 + hf * 8 + r;
      const int dcol = nj * 16 + ln;
      X[((size_t)batch * kT + trow) * kF + head * kD + dcol] =
          f2bf(accO[jj][r]);
    }
  }
}

// ---------------------------------------------------------------------------
// Kernel 2: projection GEMM  proj[r][o] = sum_f X[r][f] * W[o][f] + b[o]
//   rows R = 32*512 = 16384, K = 2048, N = 256. bf16 WMMA, f32 acc.
// ---------------------------------------------------------------------------
__global__ __launch_bounds__(256) void proj_kernel(
    const __bf16* __restrict__ X, const float* __restrict__ W,
    const float* __restrict__ bias, float* __restrict__ out) {
  __shared__ __align__(128) unsigned char smem[4096 + 32768];
  __bf16* sX = (__bf16*)smem;            // [32][64] bf16
  __bf16* sW = (__bf16*)(smem + 4096);   // [256][64] bf16
  const int r0 = blockIdx.x * 32;
  const int tid = threadIdx.x, wave = tid >> 5, lane = tid & 31;
  const int ln = lane & 15, hf = lane >> 4;
  const int mi = wave >> 2;

  v8f acc[4];
#pragma unroll
  for (int jj = 0; jj < 4; ++jj) acc[jj] = vzero8();

  for (int kc = 0; kc < 32; ++kc) {
    const int f0 = kc * 64;
    {  // stage X rows (already bf16)
      const int i = tid >> 3, fl = (tid & 7) * 8;
      *(v8bf*)(sX + i * 64 + fl) =
          *(const v8bf*)(X + (size_t)(r0 + i) * kF + f0 + fl);
    }
    {  // stage W tile (f32 -> bf16), one output row per thread
      const int o = tid;
      const float4* src = (const float4*)(W + (size_t)o * kF + f0);
#pragma unroll
      for (int j = 0; j < 16; ++j) {
        float4 f = src[j];
        int idx = o * 64 + j * 4;
        sW[idx + 0] = f2bf(f.x); sW[idx + 1] = f2bf(f.y);
        sW[idx + 2] = f2bf(f.z); sW[idx + 3] = f2bf(f.w);
      }
    }
    __syncthreads();
    const int mA = mi * 16 + ln;
#pragma unroll
    for (int kkk = 0; kkk < 2; ++kkk) {
      const int kl = kkk * 32;
      v8bf a0 = *(const v8bf*)(sX + mA * 64 + kl + hf * 8);
      v8bf a1 = *(const v8bf*)(sX + mA * 64 + kl + hf * 8 + 16);
      v16bf Af = cat16(a0, a1);
#pragma unroll
      for (int jj = 0; jj < 4; ++jj) {
        const int nj = (wave & 3) * 4 + jj;
        const int o = nj * 16 + ln;
        v8bf b0 = *(const v8bf*)(sW + o * 64 + kl + hf * 8);
        v8bf b1 = *(const v8bf*)(sW + o * 64 + kl + hf * 8 + 16);
        acc[jj] = __builtin_amdgcn_wmma_f32_16x16x32_bf16(
            false, Af, false, cat16(b0, b1), (short)0, acc[jj], false, false);
      }
    }
    __syncthreads();
  }

#pragma unroll
  for (int jj = 0; jj < 4; ++jj) {
    const int nj = (wave & 3) * 4 + jj;
    const int o = nj * 16 + ln;
    const float bo = bias[o];
#pragma unroll
    for (int r = 0; r < 8; ++r) {
      const int row = r0 + mi * 16 + hf * 8 + r;
      out[(size_t)row * kD + o] = acc[jj][r] + bo;
    }
  }
}

// ---------------------------------------------------------------------------
extern "C" void kernel_launch(void* const* d_in, const int* in_sizes, int n_in,
                              void* d_out, int out_size, void* d_ws,
                              size_t ws_size, hipStream_t stream) {
  const float* q    = (const float*)d_in[0];
  const float* k    = (const float*)d_in[1];
  const float* v    = (const float*)d_in[2];
  const float* W    = (const float*)d_in[3];
  const float* bias = (const float*)d_in[4];
  // d_in[5] = batch_size scalar (fixed 32)
  const unsigned char* mask = (const unsigned char*)d_in[6];

  float* proj = (float*)d_out;                 // [32][512][256]
  float* asum = proj + PROJ_ELEMS;             // [32][512][512]
  __bf16* X   = (__bf16*)d_ws;                 // [32][512][2048] bf16

  zero_f4<<<ASUM_ELEMS / (4 * 256), 256, 0, stream>>>((float4*)asum,
                                                      ASUM_ELEMS / 4);
  attn_kernel<<<dim3(16, 256), 256, 0, stream>>>(q, k, v, mask, X, asum);
  proj_kernel<<<(cB * kT) / 32, 256, 0, stream>>>(X, W, bias, proj);
}
